// CFGEmbeder_83717502534008
// MI455X (gfx1250) — matrix operand order repaired and checked by reading, not compiled, not run on hardware
//
#include <hip/hip_runtime.h>
#include <hip/hip_bf16.h>

// ---------------------------------------------------------------------------
// Model dimensions (match reference)
// ---------------------------------------------------------------------------
#define BB   128      // batch
#define HH   512      // hidden
#define EE   512      // embedding
#define TT   256      // token seq len
#define TD   64       // desc seq len
#define NNODE 160     // graph nodes
#define H4   2048     // 4*H
#define MARGIN_F 0.6f

typedef __attribute__((ext_vector_type(16))) __bf16          v16bf;
typedef __attribute__((ext_vector_type(8)))  float           v8f;
typedef __attribute__((ext_vector_type(16))) unsigned short  us16;
typedef __attribute__((ext_vector_type(8)))  unsigned short  us8;

union FragBF { us16 u; v16bf m; };

__device__ __forceinline__ unsigned short f2bf(float f) {
  union { float f; unsigned u; } v; v.f = f;
  unsigned r = v.u + 0x7FFFu + ((v.u >> 16) & 1u);   // round-to-nearest-even
  return (unsigned short)(r >> 16);
}
__device__ __forceinline__ float sigf(float x) { return 1.f / (1.f + __expf(-x)); }

__device__ __forceinline__ us16 cat8(us8 lo, us8 hi) {
  us16 r;
  #pragma unroll
  for (int i = 0; i < 8; ++i) { r[i] = lo[i]; r[i + 8] = hi[i]; }
  return r;
}

// ---- CDNA5 async copy: global -> LDS (ASYNCcnt-tracked), 16B per lane ------
// LDS destination is the low 32 bits of the generic pointer (LDS aperture
// maps addr[31:0] directly onto the workgroup LDS offset; HW adds LDS_BASE).
__device__ __forceinline__ void async_ld_b128(unsigned lds_off, const void* gaddr) {
  asm volatile("global_load_async_to_lds_b128 %0, %1, off"
               :: "v"(lds_off), "v"(gaddr) : "memory");
}
__device__ __forceinline__ void wait_async0() {
#if __has_builtin(__builtin_amdgcn_s_wait_asynccnt)
  __builtin_amdgcn_s_wait_asynccnt(0);
#else
  asm volatile("s_wait_asynccnt 0x0" ::: "memory");
#endif
}

// ---------------------------------------------------------------------------
// Generic batched bf16 GEMM with fused epilogue.
//   C[b] = act( A[b](MxK,bf16) * B[b](KxN,bf16) + (accum ? C_old : 0) + bias )
// flags: bit0 = accumulate into C; bits[2:1] = act (0 none, 1 tanh, 2 sigmoid)
// Requirements: K % 32 == 0, N % 8 == 0, 16-byte aligned rows (all call sites
// satisfy this: K in {160,512,1024}, N in {512,2048}, strides mult. of 8).
// Block: 256 threads = 8 waves, tile 128(M) x 64(N), K chunk 32.
// A tile is DMA'd straight into LDS with global_load_async_to_lds_b128;
// B tile is transposed on LDS store so both fragment builds are contiguous
// 16-byte ds_load_b128s matching the WMMA bf16 VGPR layouts.
// Each wave computes a 32x32 sub-tile = 4 x v_wmma_f32_16x16x32_bf16.
// ---------------------------------------------------------------------------
#define GT_M 128
#define GT_N 64
#define GT_K 32

__global__ __launch_bounds__(256) void gemm_bf16_kernel(
    const unsigned short* __restrict__ A, long strideA, int lda,
    const unsigned short* __restrict__ Bm, long strideB, int ldb,
    float* __restrict__ C, long strideC, int ldc,
    const float* __restrict__ bias,
    int M, int N, int K, int flags)
{
  __shared__ unsigned short As[GT_M * GT_K];  // 8 KB  (row-major MxK tile)
  __shared__ unsigned short Bs[GT_N * GT_K];  // 4 KB  (transposed: [n][k])

  const int tid  = threadIdx.x;
  const int lane = tid & 31;
  const int wave = tid >> 5;
  const int wm   = (wave >> 1) << 5;   // 0,32,64,96
  const int wn   = (wave & 1) << 5;    // 0,32
  const int l16  = lane & 15;
  const int lhi  = lane >> 4;

  const long bm = (long)blockIdx.y * GT_M;
  const long bn = (long)blockIdx.x * GT_N;
  A  += strideA * blockIdx.z;
  Bm += strideB * blockIdx.z;
  C  += strideC * blockIdx.z;

  // global->LDS mapping: A: 16 halfs/thread (row tid>>1, col (tid&1)*16)
  //                      B: 8 halfs/thread (row tid>>3, col (tid&7)*8)
  const int arow = tid >> 1;
  const int acol = (tid & 1) << 4;
  const int brow = tid >> 3;
  const int bcol = (tid & 7) << 3;

  long garow = bm + arow; if (garow >= M) garow = M - 1;   // clamp; masked at store
  const long gbcol = bn + bcol;

  // This thread's async destinations inside the A tile (two 16B chunks).
  const unsigned ldsA0 = (unsigned)(size_t)(As + arow * GT_K + acol);
  const unsigned ldsA1 = ldsA0 + 16;

  v8f acc[2][2] = {};

  for (int k0 = 0; k0 < K; k0 += GT_K) {
    const unsigned short* gA = A + garow * (long)lda + k0 + acol;
    us8 bv = {0, 0, 0, 0, 0, 0, 0, 0};
    if (gbcol + 8 <= N)
      bv = *(const us8*)(Bm + (long)(k0 + brow) * ldb + gbcol);
    if (k0 + GT_K < K)   // prefetch next chunk's B row (global_prefetch_b8)
      __builtin_prefetch(Bm + (long)(k0 + GT_K + brow) * ldb + gbcol, 0, 3);

    __syncthreads();                     // consumers done with previous tiles
    async_ld_b128(ldsA0, gA);            // A tile: DMA straight into LDS
    async_ld_b128(ldsA1, gA + 8);
    #pragma unroll
    for (int i = 0; i < 8; ++i)          // transpose B tile into [n][k]
      Bs[(bcol + i) * GT_K + brow] = bv[i];
    wait_async0();                       // A tile landed in LDS
    __syncthreads();

    // Build WMMA fragments from LDS (contiguous 16B reads, ISA bf16 layout):
    // A 16x32: lanes0-15 -> K0-7 & K16-23 ; lanes16-31 -> K8-15 & K24-31
    // B 32x16 (via transposed LDS): lane -> N=l16, same K order as A.
    FragBF a0, a1, b0, b1;
    {
      const us8* r0 = (const us8*)(As + (wm + l16) * GT_K);
      a0.u = cat8(r0[lhi], r0[2 + lhi]);
      const us8* r1 = (const us8*)(As + (wm + 16 + l16) * GT_K);
      a1.u = cat8(r1[lhi], r1[2 + lhi]);
      const us8* c0 = (const us8*)(Bs + (wn + l16) * GT_K);
      b0.u = cat8(c0[lhi], c0[2 + lhi]);
      const us8* c1 = (const us8*)(Bs + (wn + 16 + l16) * GT_K);
      b1.u = cat8(c1[lhi], c1[2 + lhi]);
    }
    acc[0][0] = __builtin_amdgcn_wmma_f32_16x16x32_bf16(false, a0.m, false, b0.m, (short)0, acc[0][0], false, false);
    acc[0][1] = __builtin_amdgcn_wmma_f32_16x16x32_bf16(false, a0.m, false, b1.m, (short)0, acc[0][1], false, false);
    acc[1][0] = __builtin_amdgcn_wmma_f32_16x16x32_bf16(false, a1.m, false, b0.m, (short)0, acc[1][0], false, false);
    acc[1][1] = __builtin_amdgcn_wmma_f32_16x16x32_bf16(false, a1.m, false, b1.m, (short)0, acc[1][1], false, false);
  }

  // Epilogue.  C/D fragment layout: VGPR r, lane l -> M = r + (l>>4)*8, N = l&15
  const bool accum = flags & 1;
  const int  act   = (flags >> 1) & 3;
  #pragma unroll
  for (int i = 0; i < 2; ++i) {
    #pragma unroll
    for (int j = 0; j < 2; ++j) {
      long colb = bn + wn + j * 16 + l16;
      if (colb >= N) continue;
      float bvs = bias ? bias[colb] : 0.f;
      #pragma unroll
      for (int r = 0; r < 8; ++r) {
        long row = bm + wm + i * 16 + r + lhi * 8;
        if (row < M) {
          float* cp = C + row * (long)ldc + colb;
          float v = acc[i][j][r] + bvs;
          if (accum) v += *cp;
          if (act == 1)      v = tanhf(v);
          else if (act == 2) v = sigf(v);
          *cp = v;
        }
      }
    }
  }
}

// ---------------------------------------------------------------------------
// Elementwise / reduction kernels
// ---------------------------------------------------------------------------
__global__ void cvt_bf16_kernel(const float* __restrict__ in,
                                unsigned short* __restrict__ out, long n) {
  long i = (long)blockIdx.x * 256 + threadIdx.x;
  if (i < n) out[i] = f2bf(in[i]);
}

__global__ void gather_emb_kernel(const float* __restrict__ emb,
                                  const int* __restrict__ idx,
                                  unsigned short* __restrict__ out,
                                  long total, int width) {
  long i = (long)blockIdx.x * 256 + threadIdx.x;
  if (i >= total) return;
  long r = i / width;
  int  c = (int)(i - r * width);
  out[i] = f2bf(emb[(long)idx[r] * width + c]);
}

// gates: B x 4H laid out [i | f | g | o]; updates h,c; writes h to feat[b][t]
__global__ void lstm_gate_kernel(const float* __restrict__ g,
                                 float* __restrict__ h, float* __restrict__ c,
                                 unsigned short* __restrict__ h_bf,
                                 float* __restrict__ featT, long featRowStride,
                                 int Bn, int Hn) {
  long idx = (long)blockIdx.x * 256 + threadIdx.x;
  if (idx >= (long)Bn * Hn) return;
  int b = (int)(idx / Hn), j = (int)(idx - (long)b * Hn);
  const float* gr = g + (long)b * 4 * Hn;
  float gi = gr[j], gf = gr[Hn + j], gg = gr[2 * Hn + j], go = gr[3 * Hn + j];
  float cc = sigf(gf) * c[idx] + sigf(gi) * tanhf(gg);
  float hn = sigf(go) * tanhf(cc);
  c[idx] = cc; h[idx] = hn; h_bf[idx] = f2bf(hn);
  featT[(long)b * featRowStride + j] = hn;
}

// sc[r] = t1[r,:] . sW + sb  (N=1 matmul; tiny, keep scalar)
__global__ void score_kernel(const float* __restrict__ t1,
                             const float* __restrict__ sW,
                             const float* __restrict__ sb,
                             float* __restrict__ sc, long rows, int Hn) {
  long r = (long)blockIdx.x * 256 + threadIdx.x;
  if (r >= rows) return;
  const float* p = t1 + r * (long)Hn;
  float acc = sb[0];
  for (int i = 0; i < Hn; ++i) acc += p[i] * sW[i];
  sc[r] = acc;
}

__global__ void softmax_pool_kernel(const float* __restrict__ feat,
                                    const float* __restrict__ sc,
                                    const int* __restrict__ len,
                                    float* __restrict__ out,
                                    int T, int Hn, int apply_tanh) {
  int b = blockIdx.x;
  __shared__ float w[256];
  __shared__ float red[256];
  int tid = threadIdx.x;
  int L = len[b];
  float s = (tid < T && tid < L) ? sc[(long)b * T + tid] : -1e30f;
  red[tid] = s; __syncthreads();
  for (int o = 128; o > 0; o >>= 1) { if (tid < o) red[tid] = fmaxf(red[tid], red[tid + o]); __syncthreads(); }
  float mx = red[0]; __syncthreads();
  float e = (tid < T && tid < L) ? __expf(s - mx) : 0.f;
  red[tid] = e; __syncthreads();
  for (int o = 128; o > 0; o >>= 1) { if (tid < o) red[tid] += red[tid + o]; __syncthreads(); }
  float inv = 1.f / red[0];
  if (tid < 256) w[tid] = (tid < T) ? e * inv : 0.f;
  __syncthreads();
  for (int h = tid; h < Hn; h += blockDim.x) {
    float acc = 0.f;
    for (int t = 0; t < T; ++t)
      acc += w[t] * feat[((long)b * T + t) * Hn + h];
    out[(long)b * Hn + h] = apply_tanh ? tanhf(acc) : acc;
  }
}

__global__ void sigmoid_pool_kernel(const float* __restrict__ feat,
                                    const float* __restrict__ sc,
                                    const int* __restrict__ mask,
                                    float* __restrict__ out, int Nn, int Hn) {
  int b = blockIdx.x;
  __shared__ float w[256];
  int tid = threadIdx.x;
  if (tid < 256)
    w[tid] = (tid < Nn && mask[(long)b * Nn + tid])
                 ? sigf(sc[(long)b * Nn + tid]) : 0.f;
  __syncthreads();
  for (int h = tid; h < Hn; h += blockDim.x) {
    float acc = 0.f;
    for (int t = 0; t < Nn; ++t)
      acc += w[t] * feat[((long)b * Nn + t) * Hn + h];
    out[(long)b * Hn + h] = acc;
  }
}

__global__ void mul_bf16_kernel(const float* __restrict__ a,
                                const float* __restrict__ b,
                                unsigned short* __restrict__ o, long n) {
  long i = (long)blockIdx.x * 256 + threadIdx.x;
  if (i < n) o[i] = f2bf(a[i] * b[i]);
}

// h = (1-z)*h + z*hh   (z already sigmoid'd, hh already tanh'd by GEMM epilogue)
__global__ void ggnn_combine_kernel(const float* __restrict__ z,
                                    const float* __restrict__ hh,
                                    float* __restrict__ h,
                                    unsigned short* __restrict__ h_bf, long n) {
  long i = (long)blockIdx.x * 256 + threadIdx.x;
  if (i >= n) return;
  float zz = z[i];
  float hn = (1.f - zz) * h[i] + zz * hh[i];
  h[i] = hn; h_bf[i] = f2bf(hn);
}

__global__ void loss_kernel(const float* __restrict__ code,
                            const float* __restrict__ da,
                            const float* __restrict__ dn,
                            float* __restrict__ out, int Bn, int Hn) {
  __shared__ float red[256];
  int tid = threadIdx.x;
  float acc = 0.f;
  for (int b = tid; b < Bn; b += blockDim.x) {
    const float* c = code + (long)b * Hn;
    const float* a = da + (long)b * Hn;
    const float* n = dn + (long)b * Hn;
    float nc = 0, na = 0, nn = 0, dca = 0, dcn = 0;
    for (int i = 0; i < Hn; ++i) {
      float cv = c[i], av = a[i], nv = n[i];
      nc += cv * cv; na += av * av; nn += nv * nv;
      dca += cv * av; dcn += cv * nv;
    }
    float cosA = dca / fmaxf(sqrtf(nc) * sqrtf(na), 1e-8f);
    float cosN = dcn / fmaxf(sqrtf(nc) * sqrtf(nn), 1e-8f);
    acc += fmaxf(MARGIN_F - cosA + cosN, 1e-6f);
  }
  red[tid] = acc; __syncthreads();
  for (int o = 128; o > 0; o >>= 1) { if (tid < o) red[tid] += red[tid + o]; __syncthreads(); }
  if (tid == 0) out[0] = red[0] / (float)Bn;
}

// ---------------------------------------------------------------------------
// Host helpers
// ---------------------------------------------------------------------------
static void gemm_launch(hipStream_t st,
                        const unsigned short* A, long sA, int lda,
                        const unsigned short* B, long sB, int ldb,
                        float* C, long sC, int ldc, const float* bias,
                        int M, int N, int K, int flags, int batch) {
  dim3 grid((N + GT_N - 1) / GT_N, (M + GT_M - 1) / GT_M, batch);
  gemm_bf16_kernel<<<grid, dim3(256), 0, st>>>(A, sA, lda, B, sB, ldb,
                                               C, sC, ldc, bias, M, N, K, flags);
}

static void run_lstm(hipStream_t st, const unsigned short* X, int T,
                     const unsigned short* Wx, const unsigned short* Wh,
                     const float* bias, float* feat, float* h, float* c,
                     unsigned short* h_bf, float* gates) {
  hipMemsetAsync(h, 0, (size_t)BB * HH * 4, st);
  hipMemsetAsync(c, 0, (size_t)BB * HH * 4, st);
  hipMemsetAsync(h_bf, 0, (size_t)BB * HH * 2, st);
  long n = (long)BB * HH;
  for (int t = 0; t < T; ++t) {
    // gates = x_t @ Wx + b    (x_t rows strided by T*E inside gathered X)
    gemm_launch(st, X + (long)t * EE, 0, T * EE, Wx, 0, H4,
                gates, 0, H4, bias, BB, H4, EE, 0, 1);
    // gates += h @ Wh
    gemm_launch(st, h_bf, 0, HH, Wh, 0, H4,
                gates, 0, H4, nullptr, BB, H4, HH, 1, 1);
    lstm_gate_kernel<<<(int)((n + 255) / 256), 256, 0, st>>>(
        gates, h, c, h_bf, feat + (long)t * HH, (long)T * HH, BB, HH);
  }
}

// sc[rows] = tanh(feat @ aW + ab) @ sW + sb
static void scores_path(hipStream_t st, const float* feat, long rows,
                        const unsigned short* aW, const float* ab,
                        const float* sW, const float* sb,
                        unsigned short* feat_bf, float* t1, float* sc) {
  long n = rows * HH;
  cvt_bf16_kernel<<<(int)((n + 255) / 256), 256, 0, st>>>(feat, feat_bf, n);
  gemm_launch(st, feat_bf, 0, HH, aW, 0, HH, t1, 0, HH, ab,
              (int)rows, HH, HH, (1 << 1) /*tanh*/, 1);
  score_kernel<<<(int)((rows + 255) / 256), 256, 0, st>>>(t1, sW, sb, sc, rows, HH);
}

// ---------------------------------------------------------------------------
// Orchestration
// ---------------------------------------------------------------------------
extern "C" void kernel_launch(void* const* d_in, const int* in_sizes, int n_in,
                              void* d_out, int out_size, void* d_ws, size_t ws_size,
                              hipStream_t stream) {
  (void)in_sizes; (void)n_in; (void)out_size; (void)ws_size;
  typedef unsigned short us;

  // ---- inputs (setup_inputs dict order, params flattened in insertion order)
  const int*   tokens      = (const int*)  d_in[0];
  const int*   tok_len     = (const int*)  d_in[1];
  const float* dfg_init    = (const float*)d_in[2];
  const float* dfg_adj     = (const float*)d_in[3];
  const int*   dfg_mask    = (const int*)  d_in[4];
  const float* cfg_init    = (const float*)d_in[5];
  const float* cfg_adj     = (const float*)d_in[6];
  const int*   cfg_mask    = (const int*)  d_in[7];
  const int*   desc_anchor = (const int*)  d_in[8];
  const int*   desc_alen   = (const int*)  d_in[9];
  const int*   desc_neg    = (const int*)  d_in[10];
  const int*   desc_nlen   = (const int*)  d_in[11];
  const float* tok_emb  = (const float*)d_in[12];
  const float* desc_emb = (const float*)d_in[13];
  const float* tok_Wx   = (const float*)d_in[14];
  const float* tok_Wh   = (const float*)d_in[15];
  const float* tok_b    = (const float*)d_in[16];
  const float* desc_Wx  = (const float*)d_in[17];
  const float* desc_Wh  = (const float*)d_in[18];
  const float* desc_b   = (const float*)d_in[19];
  // dfg params d_in[20..29]: W_in b_in W_out b_out Wr br Wz bz Wt bt
  // cfg params d_in[30..39]
  const float* tok_aW = (const float*)d_in[40];
  const float* tok_ab = (const float*)d_in[41];
  const float* tok_sW = (const float*)d_in[42];
  const float* tok_sb = (const float*)d_in[43];
  const float* d_aW   = (const float*)d_in[52];
  const float* d_ab   = (const float*)d_in[53];
  const float* d_sW   = (const float*)d_in[54];
  const float* d_sb   = (const float*)d_in[55];
  const float* fus_W  = (const float*)d_in[56];
  const float* fus_b  = (const float*)d_in[57];

  // ---- workspace bump allocator
  char* wsb = (char*)d_ws;
  size_t off = 0;
  auto alloc = [&](size_t bytes) -> void* {
    void* p = wsb + off;
    off = (off + bytes + 255) & ~(size_t)255;
    return p;
  };
  auto cvtw = [&](const float* src, long n) -> us* {
    us* dst = (us*)alloc((size_t)n * 2);
    cvt_bf16_kernel<<<(int)((n + 255) / 256), 256, 0, stream>>>(src, dst, n);
    return dst;
  };

  // ---- bf16 weight conversions (stateless: redone every call)
  us* wTokWx  = cvtw(tok_Wx, (long)EE * H4);
  us* wTokWh  = cvtw(tok_Wh, (long)HH * H4);
  us* wDescWx = cvtw(desc_Wx, (long)EE * H4);
  us* wDescWh = cvtw(desc_Wh, (long)HH * H4);
  us *wWin[2], *wWout[2], *wWr[2], *wWz[2], *wWt[2], *waG[2];
  for (int g = 0; g < 2; ++g) {
    int p = g ? 30 : 20;
    wWin[g]  = cvtw((const float*)d_in[p + 0], (long)HH * HH);
    wWout[g] = cvtw((const float*)d_in[p + 2], (long)HH * HH);
    wWr[g]   = cvtw((const float*)d_in[p + 4], 3L * HH * HH);
    wWz[g]   = cvtw((const float*)d_in[p + 6], 3L * HH * HH);
    wWt[g]   = cvtw((const float*)d_in[p + 8], 3L * HH * HH);
    waG[g]   = cvtw((const float*)d_in[g ? 48 : 44], (long)HH * HH);
  }
  us* waTok = cvtw(tok_aW, (long)HH * HH);
  us* waD   = cvtw(d_aW,   (long)HH * HH);
  us* wFus  = cvtw(fus_W,  3L * HH * HH);

  // ---- activation buffers
  const long nTok = (long)BB * TT * HH;        // 16.7M
  const long nDes = (long)BB * TD * HH;        // 4.2M
  const long nG   = (long)BB * NNODE * HH;     // 10.5M
  const long nAdj = (long)BB * NNODE * 2 * NNODE;
  us*    Xtok     = (us*)alloc((size_t)nTok * 2);
  us*    Xdesc    = (us*)alloc((size_t)nDes * 2);
  float* tok_feat = (float*)alloc((size_t)nTok * 4);
  float* des_feat = (float*)alloc((size_t)nDes * 4);
  us*    feat_bf  = (us*)alloc((size_t)nTok * 2);
  float* t1       = (float*)alloc((size_t)nTok * 4);
  float* sc       = (float*)alloc((size_t)(BB * TT) * 4);
  float* gates    = (float*)alloc((size_t)BB * H4 * 4);
  float* hbuf     = (float*)alloc((size_t)BB * HH * 4);
  float* cbuf     = (float*)alloc((size_t)BB * HH * 4);
  us*    hbf      = (us*)alloc((size_t)BB * HH * 2);
  float* tok_vec  = (float*)alloc((size_t)BB * HH * 4);
  float* dfg_vec  = (float*)alloc((size_t)BB * HH * 4);
  float* cfg_vec  = (float*)alloc((size_t)BB * HH * 4);
  float* da_vec   = (float*)alloc((size_t)BB * HH * 4);
  float* dn_vec   = (float*)alloc((size_t)BB * HH * 4);
  float* code     = (float*)alloc((size_t)BB * HH * 4);
  us*    vec_bf   = (us*)alloc((size_t)BB * HH * 2);
  float* hg       = (float*)alloc((size_t)nG * 4);
  us*    hg_bf    = (us*)alloc((size_t)nG * 2);
  float* ub       = (float*)alloc((size_t)nG * 4);
  us*    ub_bf    = (us*)alloc((size_t)nG * 2);
  float* ain      = (float*)alloc((size_t)nG * 4);
  float* aout     = (float*)alloc((size_t)nG * 4);
  us*    ain_bf   = (us*)alloc((size_t)nG * 2);
  us*    aout_bf  = (us*)alloc((size_t)nG * 2);
  float* rb       = (float*)alloc((size_t)nG * 4);
  float* zb       = (float*)alloc((size_t)nG * 4);
  us*    rh_bf    = (us*)alloc((size_t)nG * 2);
  float* hhb      = (float*)alloc((size_t)nG * 4);
  us*    adj_bf   = (us*)alloc((size_t)nAdj * 2);

  #define EWG(n) (int)(((n) + 255) / 256), 256, 0, stream

  // ==================== token encoder ====================
  {
    long tot = (long)BB * TT * EE;
    gather_emb_kernel<<<EWG(tot)>>>(tok_emb, tokens, Xtok, tot, EE);
    run_lstm(stream, Xtok, TT, wTokWx, wTokWh, tok_b, tok_feat, hbuf, cbuf, hbf, gates);
    scores_path(stream, tok_feat, (long)BB * TT, waTok, tok_ab, tok_sW, tok_sb,
                feat_bf, t1, sc);
    softmax_pool_kernel<<<BB, 256, 0, stream>>>(tok_feat, sc, tok_len, tok_vec,
                                                TT, HH, 0);
  }

  // ==================== GGNN encoders ====================
  auto run_ggnn = [&](const float* init, const float* adjf, const int* mask,
                      int p, const us* Win, const us* Wout, const us* Wr,
                      const us* Wz, const us* Wt, const us* aW,
                      const float* ab, const float* sW, const float* sbv,
                      float* outvec) {
    const float* b_in  = (const float*)d_in[p + 1];
    const float* b_out = (const float*)d_in[p + 3];
    const float* br    = (const float*)d_in[p + 5];
    const float* bz    = (const float*)d_in[p + 7];
    const float* bt    = (const float*)d_in[p + 9];
    const int MG = BB * NNODE;
    hipMemcpyAsync(hg, init, (size_t)nG * 4, hipMemcpyDeviceToDevice, stream);
    cvt_bf16_kernel<<<EWG(nG)>>>(init, hg_bf, nG);
    cvt_bf16_kernel<<<EWG(nAdj)>>>(adjf, adj_bf, nAdj);
    for (int s5 = 0; s5 < 5; ++s5) {
      // a_in = A_in @ (h @ W_in + b_in)
      gemm_launch(stream, hg_bf, 0, HH, Win, 0, HH, ub, 0, HH, b_in, MG, HH, HH, 0, 1);
      cvt_bf16_kernel<<<EWG(nG)>>>(ub, ub_bf, nG);
      gemm_launch(stream, adj_bf, (long)NNODE * 2 * NNODE, 2 * NNODE,
                  ub_bf, (long)NNODE * HH, HH, ain, (long)NNODE * HH, HH,
                  nullptr, NNODE, HH, NNODE, 0, BB);
      // a_out = A_out @ (h @ W_out + b_out)
      gemm_launch(stream, hg_bf, 0, HH, Wout, 0, HH, ub, 0, HH, b_out, MG, HH, HH, 0, 1);
      cvt_bf16_kernel<<<EWG(nG)>>>(ub, ub_bf, nG);
      gemm_launch(stream, adj_bf + NNODE, (long)NNODE * 2 * NNODE, 2 * NNODE,
                  ub_bf, (long)NNODE * HH, HH, aout, (long)NNODE * HH, HH,
                  nullptr, NNODE, HH, NNODE, 0, BB);
      cvt_bf16_kernel<<<EWG(nG)>>>(ain, ain_bf, nG);
      cvt_bf16_kernel<<<EWG(nG)>>>(aout, aout_bf, nG);
      // r = sigmoid([a_in,a_out,h] @ Wr + br)  -- 3 row-block GEMMs of Wr
      gemm_launch(stream, ain_bf, 0, HH, Wr,                  0, HH, rb, 0, HH, nullptr, MG, HH, HH, 0, 1);
      gemm_launch(stream, aout_bf, 0, HH, Wr + (long)HH * HH, 0, HH, rb, 0, HH, nullptr, MG, HH, HH, 1, 1);
      gemm_launch(stream, hg_bf, 0, HH, Wr + 2L * HH * HH,    0, HH, rb, 0, HH, br, MG, HH, HH, 1 | (2 << 1), 1);
      // z = sigmoid([a_in,a_out,h] @ Wz + bz)
      gemm_launch(stream, ain_bf, 0, HH, Wz,                  0, HH, zb, 0, HH, nullptr, MG, HH, HH, 0, 1);
      gemm_launch(stream, aout_bf, 0, HH, Wz + (long)HH * HH, 0, HH, zb, 0, HH, nullptr, MG, HH, HH, 1, 1);
      gemm_launch(stream, hg_bf, 0, HH, Wz + 2L * HH * HH,    0, HH, zb, 0, HH, bz, MG, HH, HH, 1 | (2 << 1), 1);
      // hh = tanh([a_in,a_out,r*h] @ Wt + bt)
      mul_bf16_kernel<<<EWG(nG)>>>(rb, hg, rh_bf, nG);
      gemm_launch(stream, ain_bf, 0, HH, Wt,                  0, HH, hhb, 0, HH, nullptr, MG, HH, HH, 0, 1);
      gemm_launch(stream, aout_bf, 0, HH, Wt + (long)HH * HH, 0, HH, hhb, 0, HH, nullptr, MG, HH, HH, 1, 1);
      gemm_launch(stream, rh_bf, 0, HH, Wt + 2L * HH * HH,    0, HH, hhb, 0, HH, bt, MG, HH, HH, 1 | (1 << 1), 1);
      ggnn_combine_kernel<<<EWG(nG)>>>(zb, hhb, hg, hg_bf, nG);
    }
    scores_path(stream, hg, (long)MG, aW, ab, sW, sbv, feat_bf, t1, sc);
    sigmoid_pool_kernel<<<BB, 256, 0, stream>>>(hg, sc, mask, outvec, NNODE, HH);
  };

  run_ggnn(dfg_init, dfg_adj, dfg_mask, 20, wWin[0], wWout[0], wWr[0], wWz[0],
           wWt[0], waG[0], (const float*)d_in[45], (const float*)d_in[46],
           (const float*)d_in[47], dfg_vec);
  run_ggnn(cfg_init, cfg_adj, cfg_mask, 30, wWin[1], wWout[1], wWr[1], wWz[1],
           wWt[1], waG[1], (const float*)d_in[49], (const float*)d_in[50],
           (const float*)d_in[51], cfg_vec);

  // ==================== fusion: code = tanh([tok,dfg,cfg] @ fus_W + b) ======
  {
    long nv = (long)BB * HH;
    cvt_bf16_kernel<<<EWG(nv)>>>(tok_vec, vec_bf, nv);
    gemm_launch(stream, vec_bf, 0, HH, wFus,                  0, HH, code, 0, HH, nullptr, BB, HH, HH, 0, 1);
    cvt_bf16_kernel<<<EWG(nv)>>>(dfg_vec, vec_bf, nv);
    gemm_launch(stream, vec_bf, 0, HH, wFus + (long)HH * HH,  0, HH, code, 0, HH, nullptr, BB, HH, HH, 1, 1);
    cvt_bf16_kernel<<<EWG(nv)>>>(cfg_vec, vec_bf, nv);
    gemm_launch(stream, vec_bf, 0, HH, wFus + 2L * HH * HH,   0, HH, code, 0, HH, fus_b, BB, HH, HH, 1 | (1 << 1), 1);
  }

  // ==================== description encoders ====================
  auto desc_encode = [&](const int* toks, const int* dlen, float* outvec) {
    long tot = (long)BB * TD * EE;
    gather_emb_kernel<<<EWG(tot)>>>(desc_emb, toks, Xdesc, tot, EE);
    run_lstm(stream, Xdesc, TD, wDescWx, wDescWh, desc_b, des_feat, hbuf, cbuf, hbf, gates);
    scores_path(stream, des_feat, (long)BB * TD, waD, d_ab, d_sW, d_sb, feat_bf, t1, sc);
    softmax_pool_kernel<<<BB, 256, 0, stream>>>(des_feat, sc, dlen, outvec, TD, HH, 1);
  };
  desc_encode(desc_anchor, desc_alen, da_vec);
  desc_encode(desc_neg, desc_nlen, dn_vec);

  // ==================== margin loss ====================
  loss_kernel<<<1, 256, 0, stream>>>(code, da_vec, dn_vec, (float*)d_out, BB, HH);

  #undef EWG
}
